// GIN_89094801588700
// MI455X (gfx1250) — matrix operand order repaired and checked by reading, not compiled, not run on hardware
//
#include <hip/hip_runtime.h>

// ---------------------------------------------------------------------------
// GIN 2-layer forward for MI455X (gfx1250, wave32, WMMA f32_16x16x32_f16)
// ---------------------------------------------------------------------------

typedef __attribute__((ext_vector_type(16))) _Float16 f16x16;
typedef __attribute__((ext_vector_type(8)))  _Float16 f16x8;
typedef __attribute__((ext_vector_type(8)))  float    f32x8;

#define HDIM 128
#define S32R 132   // padded f32 row stride (avoid LDS bank conflicts)
#define S16R 136   // padded f16 row stride
#define NEG_SLOPE 0.01f
#define LN_EPS 1e-5f

__device__ __forceinline__ f32x8 wmma_f16(f16x16 a, f16x16 b, f32x8 c) {
  return __builtin_amdgcn_wmma_f32_16x16x32_f16(
      /*neg_a=*/false, a, /*neg_b=*/false, b,
      /*c_mod=*/(short)0, c, /*reuse_a=*/false, /*reuse_b=*/false);
}

// Native fire-and-forget f32 atomic add (no return -> STOREcnt path only).
__device__ __forceinline__ void atom_add_f32(float* p, float v) {
  asm volatile("global_atomic_add_f32 %0, %1, off"
               :: "v"(p), "v"(v) : "memory");
}

// A fragment (16x32, 16-bit): lane&15 = row M; half-wave selects K subblock of 8.
// ISA 7.12.2: lanes 0-15 hold K={0..7,16..23}, lanes 16-31 hold K={8..15,24..31}.
template <typename PT>
__device__ __forceinline__ f16x16 load_a_frag(PT base, int ld, int k0) {
  const int lane = threadIdx.x & 31;
  const int r  = lane & 15;
  const int kh = (lane >> 4) << 3;          // 0 or 8
  PT p = base + r * ld + k0 + kh;
  f16x8 lo = *(const f16x8*)(p);            // K = k0+kh .. +7
  f16x8 hi = *(const f16x8*)(p + 16);       // K = k0+kh+16 .. +23
  return __builtin_shufflevector(lo, hi, 0,1,2,3,4,5,6,7,8,9,10,11,12,13,14,15);
}

// B fragment (32x16, 16-bit) sourced from row-major W[o][k] (B[k][n] = W[n][k]).
// Per ISA B layout: lane&15 = column N (= W row), lanes 0-15 hold K={0..7,16..23},
// lanes 16-31 hold K={8..15,24..31} (same as A).
template <typename PT>
__device__ __forceinline__ f16x16 load_b_frag(PT base, int ld, int k0) {
  const int lane = threadIdx.x & 31;
  const int n  = lane & 15;
  const int kh = (lane >> 4) << 3;          // 0 or 8
  PT p = base + n * ld + k0 + kh;
  f16x8 lo = *(const f16x8*)(p);
  f16x8 hi = *(const f16x8*)(p + 16);
  return __builtin_shufflevector(lo, hi, 0,1,2,3,4,5,6,7,8,9,10,11,12,13,14,15);
}

// ---------------------------------------------------------------------------
// Kernel 1: convert the 4 weight matrices to f16 (row-major [out][in])
// ---------------------------------------------------------------------------
__global__ void cvt_weights(const float* __restrict__ a, const float* __restrict__ b,
                            const float* __restrict__ c, const float* __restrict__ d,
                            _Float16* __restrict__ out) {
  const int i = blockIdx.x * blockDim.x + threadIdx.x;
  const int M = HDIM * HDIM;
  if (i < M)            out[i] = (_Float16)a[i];
  else if (i < 2 * M)   out[i] = (_Float16)b[i - M];
  else if (i < 3 * M)   out[i] = (_Float16)c[i - 2 * M];
  else if (i < 4 * M)   out[i] = (_Float16)d[i - 3 * M];
}

// ---------------------------------------------------------------------------
// Kernel 2: zero the aggregation buffer (grid-stride float4)
// ---------------------------------------------------------------------------
__global__ void zero_f32(float4* __restrict__ p, long n4) {
  long i = (long)blockIdx.x * blockDim.x + threadIdx.x;
  const long stride = (long)gridDim.x * blockDim.x;
  const float4 z = make_float4(0.f, 0.f, 0.f, 0.f);
  for (; i < n4; i += stride) p[i] = z;
}

// ---------------------------------------------------------------------------
// Kernel 3: edge scatter-add. One wave per edge: scalar endpoint loads,
// float4 gather (512 B/wave fully coalesced), 4 native f32 atomics per lane.
// ---------------------------------------------------------------------------
__global__ __launch_bounds__(256) void scatter_edges(
    const float* __restrict__ x, const int* __restrict__ src,
    const int* __restrict__ dst, float* __restrict__ agg, int E) {
  // Edge id is wave-uniform: force it scalar so src/dst become s_load.
  const int gw = __builtin_amdgcn_readfirstlane(
      (int)((blockIdx.x * (long)blockDim.x + threadIdx.x) >> 5));
  const int lane = threadIdx.x & 31;
  if (gw >= E) return;
  const int s = src[gw];
  const int d = dst[gw];
  const float4 v = ((const float4*)(x + (size_t)s * HDIM))[lane];
  float* p = agg + (size_t)d * HDIM + lane * 4;
  atom_add_f32(p + 0, v.x);
  atom_add_f32(p + 1, v.y);
  atom_add_f32(p + 2, v.z);
  atom_add_f32(p + 3, v.w);
}

// ---------------------------------------------------------------------------
// Kernel 4: fused node update:
//   t = (1+eps)*x + agg
//   u = LeakyReLU(LayerNorm(t @ W1^T + b1)) @ W2^T + b2
//   out = l2norm(u)  [optionally LeakyReLU after, for conv1]
// One wave per 16-node tile, 4 waves per block, WMMA f32_16x16x32_f16.
// ---------------------------------------------------------------------------
__global__ __launch_bounds__(128) void gin_mlp(
    const float* __restrict__ xin, const float* __restrict__ agg,
    const float* __restrict__ epsp,
    const _Float16* __restrict__ W1h, const float* __restrict__ b1,
    const float* __restrict__ gam, const float* __restrict__ bet,
    const _Float16* __restrict__ W2h, const float* __restrict__ b2,
    float* __restrict__ out, int nNodes, int postLRelu) {

  __shared__ __align__(16) float    sm32[4][16][S32R];
  __shared__ __align__(16) _Float16 sm16[4][16][S16R];
  __shared__ float smu[4][16];
  __shared__ float srs[4][16];

  const int wave = threadIdx.x >> 5;
  const int lane = threadIdx.x & 31;
  const int row0 = blockIdx.x * 64 + wave * 16;
  const float epf = 1.0f + epsp[0];

  // ---- stage input t = (1+eps)*x + agg  -> f16 in LDS -------------------
  for (int idx = lane; idx < 16 * HDIM; idx += 32) {
    const int r = idx >> 7;
    const int c = idx & (HDIM - 1);
    const int gr = row0 + r;
    float v = 0.0f;
    if (gr < nNodes) {
      const size_t o = (size_t)gr * HDIM + c;
      v = epf * xin[o] + agg[o];
    }
    sm16[wave][r][c] = (_Float16)v;
  }
  __syncthreads();

  // ---- GEMM1: (16x128) @ W1^T -> sm32, + bias1 --------------------------
  {
    f16x16 afr[4];
#pragma unroll
    for (int kk = 0; kk < 4; ++kk)
      afr[kk] = load_a_frag(&sm16[wave][0][0], S16R, kk * 32);
#pragma unroll
    for (int t = 0; t < 8; ++t) {
      f32x8 acc = {};
#pragma unroll
      for (int kk = 0; kk < 4; ++kk) {
        f16x16 bfr = load_b_frag(W1h + (size_t)t * 16 * HDIM, HDIM, kk * 32);
        acc = wmma_f16(afr[kk], bfr, acc);
      }
      const int n  = lane & 15;
      const int mb = (lane >> 4) << 3;
      const float bv = b1[t * 16 + n];
#pragma unroll
      for (int r = 0; r < 8; ++r)
        sm32[wave][mb + r][t * 16 + n] = acc[r] + bv;
    }
  }
  __syncthreads();

  // ---- LayerNorm statistics (one lane per row) --------------------------
  if (lane < 16) {
    float s = 0.f, ss = 0.f;
    const float* rp = &sm32[wave][lane][0];
#pragma unroll
    for (int c = 0; c < HDIM; c += 4) {
      const float4 v = *(const float4*)(rp + c);
      s  += v.x + v.y + v.z + v.w;
      ss += v.x * v.x + v.y * v.y + v.z * v.z + v.w * v.w;
    }
    const float mu = s * (1.0f / HDIM);
    float var = ss * (1.0f / HDIM) - mu * mu;
    var = fmaxf(var, 0.0f);
    smu[wave][lane] = mu;
    srs[wave][lane] = rsqrtf(var + LN_EPS);
  }
  __syncthreads();

  // ---- normalize + affine + LeakyReLU -> f16 ----------------------------
  for (int idx = lane; idx < 16 * HDIM; idx += 32) {
    const int r = idx >> 7;
    const int c = idx & (HDIM - 1);
    float v = sm32[wave][r][c];
    v = (v - smu[wave][r]) * srs[wave][r] * gam[c] + bet[c];
    v = (v >= 0.f) ? v : NEG_SLOPE * v;
    sm16[wave][r][c] = (_Float16)v;
  }
  __syncthreads();

  // ---- GEMM2 + bias2 -> sm32 --------------------------------------------
  {
    f16x16 afr[4];
#pragma unroll
    for (int kk = 0; kk < 4; ++kk)
      afr[kk] = load_a_frag(&sm16[wave][0][0], S16R, kk * 32);
#pragma unroll
    for (int t = 0; t < 8; ++t) {
      f32x8 acc = {};
#pragma unroll
      for (int kk = 0; kk < 4; ++kk) {
        f16x16 bfr = load_b_frag(W2h + (size_t)t * 16 * HDIM, HDIM, kk * 32);
        acc = wmma_f16(afr[kk], bfr, acc);
      }
      const int n  = lane & 15;
      const int mb = (lane >> 4) << 3;
      const float bv = b2[t * 16 + n];
#pragma unroll
      for (int r = 0; r < 8; ++r)
        sm32[wave][mb + r][t * 16 + n] = acc[r] + bv;
    }
  }
  __syncthreads();

  // ---- row L2 norm -------------------------------------------------------
  if (lane < 16) {
    float ss = 0.f;
    const float* rp = &sm32[wave][lane][0];
#pragma unroll
    for (int c = 0; c < HDIM; c += 4) {
      const float4 v = *(const float4*)(rp + c);
      ss += v.x * v.x + v.y * v.y + v.z * v.z + v.w * v.w;
    }
    srs[wave][lane] = 1.0f / fmaxf(sqrtf(ss), 1e-12f);
  }
  __syncthreads();

  // ---- scale (+ optional LeakyReLU) and write out ------------------------
  for (int idx = lane; idx < 16 * HDIM; idx += 32) {
    const int r = idx >> 7;
    const int c = idx & (HDIM - 1);
    const int gr = row0 + r;
    if (gr < nNodes) {
      float v = sm32[wave][r][c] * srs[wave][r];
      if (postLRelu) v = (v >= 0.f) ? v : NEG_SLOPE * v;
      out[(size_t)gr * HDIM + c] = v;
    }
  }
}

// ---------------------------------------------------------------------------
// Host-side launch sequencing
// ---------------------------------------------------------------------------
extern "C" void kernel_launch(void* const* d_in, const int* in_sizes, int n_in,
                              void* d_out, int out_size, void* d_ws, size_t ws_size,
                              hipStream_t stream) {
  const float* x    = (const float*)d_in[0];
  const int*   ei   = (const int*)d_in[1];
  const float* eps1 = (const float*)d_in[2];
  const float* eps2 = (const float*)d_in[3];
  const float* W1a = (const float*)d_in[4];
  const float* b1a = (const float*)d_in[5];
  const float* ga  = (const float*)d_in[6];
  const float* bea = (const float*)d_in[7];
  const float* W2a = (const float*)d_in[8];
  const float* b2a = (const float*)d_in[9];
  const float* W1b = (const float*)d_in[10];
  const float* b1b = (const float*)d_in[11];
  const float* gb  = (const float*)d_in[12];
  const float* beb = (const float*)d_in[13];
  const float* W2b = (const float*)d_in[14];
  const float* b2b = (const float*)d_in[15];

  const int N = in_sizes[0] / HDIM;
  const int E = in_sizes[1] / 2;
  const int* src = ei;
  const int* dst = ei + E;

  float* out = (float*)d_out;

  // Workspace layout: [agg: N*128 f32][weights f16: 4 * 128*128]
  float*    agg = (float*)d_ws;
  _Float16* wh  = (_Float16*)((char*)d_ws + (size_t)N * HDIM * sizeof(float));
  _Float16* W1ah = wh;
  _Float16* W2ah = wh + HDIM * HDIM;
  _Float16* W1bh = wh + 2 * HDIM * HDIM;
  _Float16* W2bh = wh + 3 * HDIM * HDIM;

  const long n4 = (long)N * HDIM / 4;
  const int scatterBlocks = (E + 7) / 8;     // 8 waves (edges) per 256-thread block
  const int mlpBlocks = (N + 63) / 64;       // 64 nodes per 128-thread block

  // Convert weights to f16 once per launch (deterministic).
  cvt_weights<<<(4 * HDIM * HDIM + 255) / 256, 256, 0, stream>>>(W1a, W2a, W1b, W2b, wh);

  // ---- conv 1 ----
  zero_f32<<<2048, 256, 0, stream>>>((float4*)agg, n4);
  scatter_edges<<<scatterBlocks, 256, 0, stream>>>(x, src, dst, agg, E);
  gin_mlp<<<mlpBlocks, 128, 0, stream>>>(x, agg, eps1, W1ah, b1a, ga, bea, W2ah, b2a,
                                         out, N, /*postLRelu=*/1);  // h1 -> d_out

  // ---- conv 2 (h1 lives in d_out; per-row read-then-write is block-local) ----
  zero_f32<<<2048, 256, 0, stream>>>((float4*)agg, n4);
  scatter_edges<<<scatterBlocks, 256, 0, stream>>>(out, src, dst, agg, E);
  gin_mlp<<<mlpBlocks, 128, 0, stream>>>(out, agg, eps2, W1bh, b1b, gb, beb, W2bh, b2b,
                                         out, N, /*postLRelu=*/0);
}
